// GraphConv_80522046865713
// MI455X (gfx1250) — matrix operand order repaired and checked by reading, not compiled, not run on hardware
//
#include <hip/hip_runtime.h>
#include <hip/hip_bf16.h>

// ---------------------------------------------------------------------------
// Graph attention for MI455X (gfx1250, wave32, WMMA).
// B=16, N=2048, D=64. Output = y [B,N,64] ++ attn [B,N,N] (fp32).
// Bandwidth-bound: adj read (268MB, once) + attn write (268MB, once).
// K/Q/V (12MB f16) are L2-resident; scores recomputed in pass 2 (WMMA ~free).
// All WMMA operands are contiguous-per-lane vector loads (no scalar packing):
//   - B-operands: 16 contiguous halfs at fixed n (K row-major, Vt d-major,
//     weights pre-transposed).
//   - A-operands: two contiguous v8h chunks per lane.
// ---------------------------------------------------------------------------

typedef _Float16 half_t;
typedef __attribute__((ext_vector_type(16))) _Float16 v16h;
typedef __attribute__((ext_vector_type(8)))  _Float16 v8h;
typedef __attribute__((ext_vector_type(8)))  float    v8f;

#define GA_B 16
#define GA_N 2048
#define GA_D 64
#define GA_BN (GA_B * GA_N)          // 32768 rows
#define NEG_INF_F (-1.0e9f)

static __device__ __forceinline__ v8f wmma_f16(v16h a, v16h b, v8f c) {
  return __builtin_amdgcn_wmma_f32_16x16x32_f16(false, a, false, b, (short)0, c,
                                                false, false);
}

// A-operand 16x32 slice at k0 from row-major row pointer (row = this lane's M).
// Lane layout: halfs [k0+16h .. k0+16h+15] (contiguous).
static __device__ __forceinline__ v16h load_A16(const half_t* row, int k0,
                                                int hh) {
  v8h lo = *(const v8h*)(row + k0 + 16 * hh);
  v8h hi = *(const v8h*)(row + k0 + 16 * hh + 8);
  return __builtin_shufflevector(lo, hi, 0, 1, 2, 3, 4, 5, 6, 7, 8, 9, 10, 11,
                                 12, 13, 14, 15);
}

// B-operand 32x16: 16 contiguous halfs (caller applies n, k0, 16*hh offsets).
static __device__ __forceinline__ v16h load_B16(const half_t* base) {
  return *(const v16h*)base;
}

#define WAIT_DS() asm volatile("s_wait_dscnt 0" ::: "memory")

// Reduce across the 16 lanes that share a C/D-matrix row.
static __device__ __forceinline__ float rowmax16(float v) {
#pragma unroll
  for (int m = 1; m < 16; m <<= 1) v = fmaxf(v, __shfl_xor(v, m, 32));
  return v;
}
static __device__ __forceinline__ float rowsum16(float v) {
#pragma unroll
  for (int m = 1; m < 16; m <<= 1) v += __shfl_xor(v, m, 32);
  return v;
}

// ---------------------------------------------------------------------------
__global__ void ga_cast_f16(const float* __restrict__ src,
                            half_t* __restrict__ dst, int n) {
  int i = blockIdx.x * blockDim.x + threadIdx.x;
  int stride = gridDim.x * blockDim.x;
  for (; i < n; i += stride) dst[i] = (half_t)src[i];
}

// Cast + transpose weight: src [K][Nn] f32 -> dst [Nn][K] f16 (n-major).
__global__ void ga_cast_t(const float* __restrict__ src,
                          half_t* __restrict__ dst, int K, int Nn) {
  int i = blockIdx.x * blockDim.x + threadIdx.x;
  if (i < K * Nn) {
    int k = i / Nn, n = i - k * Nn;
    dst[n * K + k] = (half_t)src[i];
  }
}

// ---------------------------------------------------------------------------
// Q/K/V = x @ W + b. One wave per 16-row tile. Q,K row-major f16; V d-major.
__global__ void ga_qkv(const half_t* __restrict__ Xh,
                       const half_t* __restrict__ Wqt,
                       const half_t* __restrict__ Wkt,
                       const half_t* __restrict__ Wvt,
                       const float* __restrict__ bq,
                       const float* __restrict__ bk,
                       const float* __restrict__ bv,
                       half_t* __restrict__ Qh, half_t* __restrict__ Kh,
                       half_t* __restrict__ Vt) {
  const int lane = threadIdx.x;
  const int nn = lane & 15, hh = lane >> 4;
  const size_t rowbase = (size_t)blockIdx.x * 16;

  const half_t* xrow = Xh + (rowbase + nn) * GA_D;
  const v16h a0 = load_A16(xrow, 0, hh);
  const v16h a1 = load_A16(xrow, 32, hh);

  // Q and K: row-major outputs.
  const half_t* Wt[2] = {Wqt, Wkt};
  const float* bs[2] = {bq, bk};
  half_t* Os[2] = {Qh, Kh};
#pragma unroll
  for (int o = 0; o < 2; ++o) {
    const half_t* wn = Wt[o] + nn * GA_D;
#pragma unroll
    for (int nt = 0; nt < 4; ++nt) {
      // output col = nt*16 + nn ; weight col-block via transposed layout
      const half_t* wb = Wt[o] + (nt * 16 + nn) * GA_D;
      v8f acc = {};
      acc = wmma_f16(a0, load_B16(wb + 16 * hh), acc);
      acc = wmma_f16(a1, load_B16(wb + 32 + 16 * hh), acc);
      const float bias = bs[o][nt * 16 + nn];
#pragma unroll
      for (int v = 0; v < 8; ++v)
        Os[o][(rowbase + v + 8 * hh) * GA_D + nt * 16 + nn] =
            (half_t)(acc[v] + bias);
    }
    (void)wn;
  }

  // V: d-major output Vt[b][d][row] -> packed b128 stores (8 rows per lane).
  const int b = (int)(rowbase >> 11);             // rowbase / N
  const int rowb0 = ((int)rowbase & (GA_N - 1)) + 8 * hh;
#pragma unroll
  for (int nt = 0; nt < 4; ++nt) {
    const half_t* wb = Wvt + (nt * 16 + nn) * GA_D;
    v8f acc = {};
    acc = wmma_f16(a0, load_B16(wb + 16 * hh), acc);
    acc = wmma_f16(a1, load_B16(wb + 32 + 16 * hh), acc);
    const float bias = bv[nt * 16 + nn];
    v8h pk;
#pragma unroll
    for (int v = 0; v < 8; ++v) pk[v] = (half_t)(acc[v] + bias);
    *(v8h*)(Vt + ((size_t)b * GA_D + nt * 16 + nn) * GA_N + rowb0) = pk;
  }
}

// ---------------------------------------------------------------------------
// Fused masked-softmax attention. One wave per (batch, 16-row tile).
// Pass 1: WMMA scores, per-lane online softmax, adj mask bits cached in VGPRs.
// Pass 2: recompute scores, write normalized attn once, fuse ctx += P @ V.
__global__ void ga_attn(const half_t* __restrict__ Qh,
                        const half_t* __restrict__ Kh,
                        const half_t* __restrict__ Vt,
                        const int* __restrict__ adj,
                        float* __restrict__ attn,
                        half_t* __restrict__ Ctxh) {
  __shared__ alignas(16) half_t pt[16 * 32];  // P tile, A-layout source (1KB)

  const int lane = threadIdx.x;
  const int nn = lane & 15, hh = lane >> 4;
  const int b = blockIdx.x >> 7;              // 128 row tiles per batch
  const int rowbase = (blockIdx.x & 127) << 4;
  const size_t growbase = (size_t)b * GA_N + rowbase;
  const size_t kbase = (size_t)b * GA_N * GA_D;   // K row-major (halfs)
  const size_t vtbase = (size_t)b * GA_D * GA_N;  // V d-major (halfs)
  const size_t adjrow0 = (size_t)b * GA_N * GA_N + (size_t)rowbase * GA_N;
  const float scale = 0.125f;                 // 1/sqrt(64)

  // Q A-operands straight from global (L2-resident).
  const half_t* qrow = Qh + (growbase + nn) * GA_D;
  const v16h aq0 = load_A16(qrow, 0, hh);
  const v16h aq1 = load_A16(qrow, 32, hh);

  float mrun[8], lrun[8], il[8];
  unsigned mask[8][4];
#pragma unroll
  for (int v = 0; v < 8; ++v) { mrun[v] = -3.0e38f; lrun[v] = 0.0f; }

  // ---- Pass 1: per-lane online softmax + mask capture (adj read ONCE) ----
#pragma unroll
  for (int w = 0; w < 4; ++w) {
    unsigned mw[8] = {};
#pragma unroll 1
    for (int t = 0; t < 32; ++t) {
      const int j0 = ((w << 5) + t) << 4;
      const half_t* kb = Kh + kbase + (size_t)(j0 + nn) * GA_D;
      v8f s = {};
      s = wmma_f16(aq0, load_B16(kb + 16 * hh), s);
      s = wmma_f16(aq1, load_B16(kb + 32 + 16 * hh), s);
#pragma unroll
      for (int v = 0; v < 8; ++v) {
        const int m = v + 8 * hh;
        const int a = adj[adjrow0 + (size_t)m * GA_N + j0 + nn];
        const unsigned bit = (a > 0) ? 1u : 0u;
        mw[v] |= bit << t;
        const float sm = bit ? s[v] * scale : NEG_INF_F;
        const float nm = fmaxf(mrun[v], sm);
        lrun[v] = lrun[v] * __expf(mrun[v] - nm) + __expf(sm - nm);
        mrun[v] = nm;
      }
    }
#pragma unroll
    for (int v = 0; v < 8; ++v) mask[v][w] = mw[v];
  }

  // Merge per-lane stats across the 16 lanes of each row (once).
#pragma unroll
  for (int v = 0; v < 8; ++v) {
    const float mrow = rowmax16(mrun[v]);
    const float lrow = rowsum16(lrun[v] * __expf(mrun[v] - mrow));
    mrun[v] = mrow;
    il[v] = 1.0f / lrow;
  }

  v8f ctx[4];
  {
    v8f z = {};
#pragma unroll
    for (int nt = 0; nt < 4; ++nt) ctx[nt] = z;
  }

  // ---- Pass 2: recompute, normalize, write attn ONCE, fuse ctx = P@V ----
#pragma unroll
  for (int w = 0; w < 4; ++w) {
    unsigned mw[8];
#pragma unroll
    for (int v = 0; v < 8; ++v) mw[v] = mask[v][w];
#pragma unroll 1
    for (int g = 0; g < 16; ++g) {
      const int j2 = (w << 9) + (g << 5);     // 32-column group
#pragma unroll
      for (int st = 0; st < 2; ++st) {
        const int j0 = j2 + (st << 4);
        const int t = (g << 1) + st;
        const half_t* kb = Kh + kbase + (size_t)(j0 + nn) * GA_D;
        v8f s = {};
        s = wmma_f16(aq0, load_B16(kb + 16 * hh), s);
        s = wmma_f16(aq1, load_B16(kb + 32 + 16 * hh), s);
#pragma unroll
        for (int v = 0; v < 8; ++v) {
          const int m = v + 8 * hh;
          const unsigned bit = (mw[v] >> t) & 1u;
          const float sm = bit ? s[v] * scale : NEG_INF_F;
          const float p = __expf(sm - mrun[v]) * il[v];
          attn[(growbase + m) * GA_N + j0 + nn] = p;
          pt[m * 32 + (st << 4) + nn] = (half_t)p;
        }
      }
      WAIT_DS();
      const v16h ap = load_A16(pt + nn * 32, 0, hh);  // P: 16 rows x 32 keys
#pragma unroll
      for (int nt = 0; nt < 4; ++nt) {
        const half_t* vb =
            Vt + vtbase + (size_t)(nt * 16 + nn) * GA_N + j2 + 16 * hh;
        ctx[nt] = wmma_f16(ap, load_B16(vb), ctx[nt]);
      }
    }
  }

#pragma unroll
  for (int nt = 0; nt < 4; ++nt)
#pragma unroll
    for (int v = 0; v < 8; ++v)
      Ctxh[(growbase + v + 8 * hh) * GA_D + nt * 16 + nn] =
          (half_t)ctx[nt][v];
}

// ---------------------------------------------------------------------------
// o = ctx@Wo + bo ; y = relu(x@Wl[0:64] + o@Wl[64:128] + bl)
__global__ void ga_out(const half_t* __restrict__ Xh,
                       const half_t* __restrict__ Ctxh,
                       const half_t* __restrict__ Wot,
                       const float* __restrict__ bo,
                       const half_t* __restrict__ Wlt,  // [64][128] n-major
                       const float* __restrict__ bl,
                       float* __restrict__ y) {
  __shared__ alignas(16) half_t ot[16 * GA_D];  // o tile transpose (2KB)
  const int lane = threadIdx.x;
  const int nn = lane & 15, hh = lane >> 4;
  const size_t rowbase = (size_t)blockIdx.x * 16;

  const half_t* crow = Ctxh + (rowbase + nn) * GA_D;
  const v16h ac0 = load_A16(crow, 0, hh);
  const v16h ac1 = load_A16(crow, 32, hh);

#pragma unroll
  for (int nt = 0; nt < 4; ++nt) {
    const half_t* wb = Wot + (nt * 16 + nn) * GA_D;
    v8f acc = {};
    acc = wmma_f16(ac0, load_B16(wb + 16 * hh), acc);
    acc = wmma_f16(ac1, load_B16(wb + 32 + 16 * hh), acc);
    const float bias = bo[nt * 16 + nn];
#pragma unroll
    for (int v = 0; v < 8; ++v)
      ot[(v + 8 * hh) * GA_D + nt * 16 + nn] = (half_t)(acc[v] + bias);
  }
  WAIT_DS();

  const half_t* xrow = Xh + (rowbase + nn) * GA_D;
  const v16h ax0 = load_A16(xrow, 0, hh);
  const v16h ax1 = load_A16(xrow, 32, hh);
  const v16h ao0 = load_A16(ot + nn * GA_D, 0, hh);
  const v16h ao1 = load_A16(ot + nn * GA_D, 32, hh);

#pragma unroll
  for (int nt = 0; nt < 4; ++nt) {
    const half_t* wb = Wlt + (nt * 16 + nn) * 128;  // rows of Wl^T, len 128
    v8f acc = {};
    acc = wmma_f16(ax0, load_B16(wb + 16 * hh), acc);
    acc = wmma_f16(ax1, load_B16(wb + 32 + 16 * hh), acc);
    acc = wmma_f16(ao0, load_B16(wb + 64 + 16 * hh), acc);
    acc = wmma_f16(ao1, load_B16(wb + 96 + 16 * hh), acc);
    const float bias = bl[nt * 16 + nn];
#pragma unroll
    for (int v = 0; v < 8; ++v)
      y[(rowbase + v + 8 * hh) * GA_D + nt * 16 + nn] =
          fmaxf(acc[v] + bias, 0.0f);
  }
}

// ---------------------------------------------------------------------------
extern "C" void kernel_launch(void* const* d_in, const int* in_sizes, int n_in,
                              void* d_out, int out_size, void* d_ws,
                              size_t ws_size, hipStream_t stream) {
  const float* x  = (const float*)d_in[0];
  const int*  adj = (const int*)d_in[1];
  const float* Wq = (const float*)d_in[2];
  const float* bq = (const float*)d_in[3];
  const float* Wk = (const float*)d_in[4];
  const float* bk = (const float*)d_in[5];
  const float* Wv = (const float*)d_in[6];
  const float* bv = (const float*)d_in[7];
  const float* Wo = (const float*)d_in[8];
  const float* bo = (const float*)d_in[9];
  const float* Wl = (const float*)d_in[10];
  const float* bl = (const float*)d_in[11];

  // Workspace layout (f16 tensors), ~20 MB total.
  char* p = (char*)d_ws;
  const size_t tsz = (size_t)GA_BN * GA_D * sizeof(half_t);  // 4 MB
  half_t* Xh   = (half_t*)p; p += tsz;
  half_t* Qh   = (half_t*)p; p += tsz;
  half_t* Kh   = (half_t*)p; p += tsz;
  half_t* Vt   = (half_t*)p; p += tsz;   // d-major V
  half_t* Ctxh = (half_t*)p; p += tsz;
  half_t* Wqt  = (half_t*)p; p += GA_D * GA_D * sizeof(half_t);
  half_t* Wkt  = (half_t*)p; p += GA_D * GA_D * sizeof(half_t);
  half_t* Wvt  = (half_t*)p; p += GA_D * GA_D * sizeof(half_t);
  half_t* Wot  = (half_t*)p; p += GA_D * GA_D * sizeof(half_t);
  half_t* Wlt  = (half_t*)p; p += 2 * GA_D * GA_D * sizeof(half_t);

  float* y_out    = (float*)d_out;
  float* attn_out = y_out + (size_t)GA_BN * GA_D;

  ga_cast_f16<<<512, 256, 0, stream>>>(x, Xh, GA_BN * GA_D);
  ga_cast_t<<<16, 256, 0, stream>>>(Wq, Wqt, GA_D, GA_D);
  ga_cast_t<<<16, 256, 0, stream>>>(Wk, Wkt, GA_D, GA_D);
  ga_cast_t<<<16, 256, 0, stream>>>(Wv, Wvt, GA_D, GA_D);
  ga_cast_t<<<16, 256, 0, stream>>>(Wo, Wot, GA_D, GA_D);
  ga_cast_t<<<32, 256, 0, stream>>>(Wl, Wlt, 2 * GA_D, GA_D);

  const int ntiles = GA_BN / 16;  // 2048 16-row tiles
  ga_qkv<<<ntiles, 32, 0, stream>>>(Xh, Wqt, Wkt, Wvt, bq, bk, bv, Qh, Kh, Vt);
  ga_attn<<<ntiles, 32, 0, stream>>>(Qh, Kh, Vt, adj, attn_out, Ctxh);
  ga_out<<<ntiles, 32, 0, stream>>>(Xh, Ctxh, Wot, bo, Wlt, bl, y_out);
}
